// WeightDropLSTM_73272142069831
// MI455X (gfx1250) — compile-verified, matrix operand order checked by reading, not run on hardware
//
#include <hip/hip_runtime.h>
#include <cstdint>

typedef __attribute__((ext_vector_type(16))) __bf16 v16bf;
typedef __attribute__((ext_vector_type(8)))  __bf16 v8bf;
typedef __attribute__((ext_vector_type(8)))  float  v8f;

#define HDIM 1024
#define KDIM 2048          // I + H concatenated reduction dim
#define BM   128           // batch rows per block
#define BK   32            // K-step (matches 16x16x32 bf16 WMMA)
#define LDST 40            // BK + 8 pad -> 80B row stride, conflict-free ds_b128

__device__ __forceinline__ v8bf cvt8(const float4 a, const float4 b) {
    v8bf r;
    r[0]=(__bf16)a.x; r[1]=(__bf16)a.y; r[2]=(__bf16)a.z; r[3]=(__bf16)a.w;
    r[4]=(__bf16)b.x; r[5]=(__bf16)b.y; r[6]=(__bf16)b.z; r[7]=(__bf16)b.w;
    return r;
}

__device__ __forceinline__ float sigm(float x)      { return 1.f/(1.f+__expf(-x)); }
__device__ __forceinline__ float tanh_fast(float x) { return 2.f/(1.f+__expf(-2.f*x)) - 1.f; }

// ---------------------------------------------------------------------------
// Pack pass: concatenate two f32 sources row-wise into one bf16 matrix
// out[row][0:1024) = s0[row], out[row][1024:2048) = s1[row]
// ---------------------------------------------------------------------------
__global__ __launch_bounds__(256)
void pack_bf16_2src(const float* __restrict__ s0, const float* __restrict__ s1,
                    __bf16* __restrict__ out)
{
    const size_t i   = ((size_t)blockIdx.x * 256 + threadIdx.x) * 8;
    const size_t row = i >> 11;                 // / KDIM
    const int    k   = (int)(i & (KDIM - 1));
    const float* src = (k < HDIM) ? (s0 + row*HDIM + k) : (s1 + row*HDIM + (k - HDIM));
    float4 a0 = ((const float4*)src)[0];
    float4 a1 = ((const float4*)src)[1];
    *(v8bf*)(out + i) = cvt8(a0, a1);
}

// ---------------------------------------------------------------------------
// Main GEMM: bf16 A[B,2048] x W[4096,2048]^T via WMMA, async-to-LDS staging,
// fused LSTM epilogue. Block = 128 batch rows x 32 h-cols x all 4 gates.
// ---------------------------------------------------------------------------
__global__ __launch_bounds__(256)
void lstm_gemm_async(const __bf16* __restrict__ Abf, const __bf16* __restrict__ Wbf,
                     const float* __restrict__ cx,   const float* __restrict__ bias,
                     float* __restrict__ hy, float* __restrict__ cy)
{
    __shared__ __align__(16) __bf16 As[2][BM*LDST];
    __shared__ __align__(16) __bf16 Ws[2][BM*LDST];

    const int h0 = blockIdx.x * 32;
    const int m0 = blockIdx.y * BM;

    const int t     = threadIdx.x;
    const int wave  = t >> 5;
    const int waveM = wave & 3;        // *32 batch rows
    const int hcSel = wave >> 2;       // *16 h columns
    const int lane  = t & 31;
    const int sel   = lane >> 4;
    const int lr    = lane & 15;

    // async staging: thread owns 32B of one tile row (two b128 transfers each)
    const int srow = t >> 1;           // 0..127
    const int scb  = (t & 1) * 32;     // byte col {0, 32} of the 64B bf16 row

    const __bf16* aRow = Abf + (size_t)(m0 + srow) * KDIM + (scb >> 1);
    const int     wn   = (srow >> 5) * HDIM + h0 + (srow & 31);   // gate*1024 + h
    const __bf16* wRow = Wbf + (size_t)wn * KDIM + (scb >> 1);

    auto issue = [&](int bufSel, int kb) {
        const __bf16* ga = aRow + kb;
        const __bf16* gw = wRow + kb;
        const uint32_t la = (uint32_t)(uintptr_t)(&As[bufSel][srow * LDST]) + scb;
        const uint32_t lw = (uint32_t)(uintptr_t)(&Ws[bufSel][srow * LDST]) + scb;
        // offset:16 applies to BOTH the global and LDS address (ISA 10. async ops)
        asm volatile("global_load_async_to_lds_b128 %0, %1, off"           :: "v"(la), "v"(ga) : "memory");
        asm volatile("global_load_async_to_lds_b128 %0, %1, off offset:16":: "v"(la), "v"(ga) : "memory");
        asm volatile("global_load_async_to_lds_b128 %0, %1, off"           :: "v"(lw), "v"(gw) : "memory");
        asm volatile("global_load_async_to_lds_b128 %0, %1, off offset:16":: "v"(lw), "v"(gw) : "memory");
    };

    v8f acc[2][4] = {};                 // [m-tile][gate] 16x16 f32 accums

    issue(0, 0);                        // ASYNCcnt = 4
    int buf = 0;
#pragma unroll 1
    for (int kb = 0; kb < KDIM; kb += BK) {
        if (kb + BK < KDIM) {
            issue(buf ^ 1, kb + BK);                      // ASYNCcnt -> 8
            asm volatile("s_wait_asynccnt 0x4" ::: "memory");  // this stage's 4 landed
        } else {
            asm volatile("s_wait_asynccnt 0x0" ::: "memory");
        }
        __syncthreads();               // all threads' async data visible

        v16bf af[2], bfr[4];
#pragma unroll
        for (int mt = 0; mt < 2; ++mt) {
            const __bf16* p = &As[buf][(waveM*32 + mt*16 + lr)*LDST + sel*8];
            v8bf lo = *(const v8bf*)p;          // K = sel*8 .. +7
            v8bf hi = *(const v8bf*)(p + 16);   // K = 16+sel*8 .. +7
            af[mt] = __builtin_shufflevector(lo, hi, 0,1,2,3,4,5,6,7,8,9,10,11,12,13,14,15);
        }
#pragma unroll
        for (int g = 0; g < 4; ++g) {
            const __bf16* p = &Ws[buf][(g*32 + hcSel*16 + lr)*LDST + sel*8];
            v8bf lo = *(const v8bf*)p;
            v8bf hi = *(const v8bf*)(p + 16);
            bfr[g] = __builtin_shufflevector(lo, hi, 0,1,2,3,4,5,6,7,8,9,10,11,12,13,14,15);
        }
#pragma unroll
        for (int mt = 0; mt < 2; ++mt)
#pragma unroll
            for (int g = 0; g < 4; ++g)
                acc[mt][g] = __builtin_amdgcn_wmma_f32_16x16x32_bf16(
                    false, af[mt], false, bfr[g], (short)0, acc[mt][g], false, false);

        __syncthreads();               // reads of buf done before next overwrite
        buf ^= 1;
    }

    // fused epilogue: this wave holds i,f,g,o for its (m, h) tile in registers
    const int hg  = h0 + hcSel*16 + lr;
    const float bi = bias[0*HDIM + hg];
    const float bfv= bias[1*HDIM + hg];
    const float bg = bias[2*HDIM + hg];
    const float bo = bias[3*HDIM + hg];

#pragma unroll
    for (int mt = 0; mt < 2; ++mt) {
#pragma unroll
        for (int j = 0; j < 8; ++j) {
            const int m = m0 + waveM*32 + mt*16 + sel*8 + j;
            const size_t idx = (size_t)m*HDIM + hg;
            const float iv = sigm(acc[mt][0][j] + bi);
            const float fv = sigm(acc[mt][1][j] + bfv);
            const float gv = tanh_fast(acc[mt][2][j] + bg);
            const float ov = sigm(acc[mt][3][j] + bo);
            const float c  = fv * cx[idx] + iv * gv;
            hy[idx] = ov * tanh_fast(c);
            cy[idx] = c;
        }
    }
}

// ---------------------------------------------------------------------------
// Fallback (no workspace): fused f32->bf16 conversion during LDS staging
// ---------------------------------------------------------------------------
__global__ __launch_bounds__(256)
void lstm_cell_wmma_fused(const float* __restrict__ inp, const float* __restrict__ hx,
                          const float* __restrict__ cx,  const float* __restrict__ wih,
                          const float* __restrict__ whh, const float* __restrict__ bias,
                          float* __restrict__ hy, float* __restrict__ cy)
{
    __shared__ __align__(16) __bf16 As[2][BM*LDST];
    __shared__ __align__(16) __bf16 Ws[2][BM*LDST];

    const int h0 = blockIdx.x * 32;
    const int m0 = blockIdx.y * BM;

    const int t     = threadIdx.x;
    const int wave  = t >> 5;
    const int waveM = wave & 3;
    const int hcSel = wave >> 2;
    const int lane  = t & 31;
    const int sel   = lane >> 4;
    const int lr    = lane & 15;

    const int srow = t >> 2;
    const int scol = (t & 3) * 8;

    auto stage = [&](int buf, int kb) {
#pragma unroll
        for (int rr = 0; rr < 2; ++rr) {
            const int row = srow + rr*64;
            const int k   = kb + scol;
            const float* ap = (k < HDIM) ? (inp + (size_t)(m0+row)*HDIM + k)
                                         : (hx  + (size_t)(m0+row)*HDIM + (k-HDIM));
            float4 a0 = ((const float4*)ap)[0];
            float4 a1 = ((const float4*)ap)[1];
            *(v8bf*)&As[buf][row*LDST + scol] = cvt8(a0, a1);
            const int n = (row >> 5)*HDIM + h0 + (row & 31);
            const float* wp = (k < HDIM) ? (wih + (size_t)n*HDIM + k)
                                         : (whh + (size_t)n*HDIM + (k-HDIM));
            float4 w0 = ((const float4*)wp)[0];
            float4 w1 = ((const float4*)wp)[1];
            *(v8bf*)&Ws[buf][row*LDST + scol] = cvt8(w0, w1);
        }
    };

    v8f acc[2][4] = {};
    stage(0, 0);
    __syncthreads();

    int buf = 0;
#pragma unroll 1
    for (int kb = 0; kb < KDIM; kb += BK) {
        if (kb + BK < KDIM) stage(buf ^ 1, kb + BK);

        v16bf af[2], bfr[4];
#pragma unroll
        for (int mt = 0; mt < 2; ++mt) {
            const __bf16* p = &As[buf][(waveM*32 + mt*16 + lr)*LDST + sel*8];
            v8bf lo = *(const v8bf*)p;
            v8bf hi = *(const v8bf*)(p + 16);
            af[mt] = __builtin_shufflevector(lo, hi, 0,1,2,3,4,5,6,7,8,9,10,11,12,13,14,15);
        }
#pragma unroll
        for (int g = 0; g < 4; ++g) {
            const __bf16* p = &Ws[buf][(g*32 + hcSel*16 + lr)*LDST + sel*8];
            v8bf lo = *(const v8bf*)p;
            v8bf hi = *(const v8bf*)(p + 16);
            bfr[g] = __builtin_shufflevector(lo, hi, 0,1,2,3,4,5,6,7,8,9,10,11,12,13,14,15);
        }
#pragma unroll
        for (int mt = 0; mt < 2; ++mt)
#pragma unroll
            for (int g = 0; g < 4; ++g)
                acc[mt][g] = __builtin_amdgcn_wmma_f32_16x16x32_bf16(
                    false, af[mt], false, bfr[g], (short)0, acc[mt][g], false, false);

        __syncthreads();
        buf ^= 1;
    }

    const int hg  = h0 + hcSel*16 + lr;
    const float bi = bias[0*HDIM + hg];
    const float bfv= bias[1*HDIM + hg];
    const float bg = bias[2*HDIM + hg];
    const float bo = bias[3*HDIM + hg];

#pragma unroll
    for (int mt = 0; mt < 2; ++mt) {
#pragma unroll
        for (int j = 0; j < 8; ++j) {
            const int m = m0 + waveM*32 + mt*16 + sel*8 + j;
            const size_t idx = (size_t)m*HDIM + hg;
            const float iv = sigm(acc[mt][0][j] + bi);
            const float fv = sigm(acc[mt][1][j] + bfv);
            const float gv = tanh_fast(acc[mt][2][j] + bg);
            const float ov = sigm(acc[mt][3][j] + bo);
            const float c  = fv * cx[idx] + iv * gv;
            hy[idx] = ov * tanh_fast(c);
            cy[idx] = c;
        }
    }
}

extern "C" void kernel_launch(void* const* d_in, const int* in_sizes, int n_in,
                              void* d_out, int out_size, void* d_ws, size_t ws_size,
                              hipStream_t stream) {
    const float* inp  = (const float*)d_in[0];
    const float* hx   = (const float*)d_in[1];
    const float* cx   = (const float*)d_in[2];
    const float* wih  = (const float*)d_in[3];
    const float* whh  = (const float*)d_in[4];
    const float* bias = (const float*)d_in[5];

    const int Bsz = in_sizes[0] / HDIM;              // 8192
    float* hy = (float*)d_out;
    float* cy = hy + (size_t)Bsz * HDIM;             // tuple (hy, cy) flat-concat

    const size_t needA = (size_t)Bsz * KDIM * sizeof(__bf16);        // 32 MB
    const size_t needW = (size_t)4 * HDIM * KDIM * sizeof(__bf16);   // 16 MB
    dim3 grid(HDIM / 32, Bsz / BM);                  // (32, 64)

    if (ws_size >= needA + needW) {
        __bf16* Abf = (__bf16*)d_ws;
        __bf16* Wbf = (__bf16*)((char*)d_ws + needA);
        pack_bf16_2src<<<(int)((size_t)Bsz * KDIM / 2048), 256, 0, stream>>>(inp, hx, Abf);
        pack_bf16_2src<<<(int)((size_t)4 * HDIM * KDIM / 2048), 256, 0, stream>>>(wih, whh, Wbf);
        lstm_gemm_async<<<grid, 256, 0, stream>>>(Abf, Wbf, cx, bias, hy, cy);
    } else {
        lstm_cell_wmma_fused<<<grid, 256, 0, stream>>>(inp, hx, cx, wih, whh, bias, hy, cy);
    }
}
